// PatchQwen3MoeExperts_3959959847401
// MI455X (gfx1250) — matrix operand order repaired and checked by reading, not compiled, not run on hardware
//
#include <hip/hip_runtime.h>

// Problem constants (from reference)
#define T_TOK 2048
#define KSEL  8
#define NEXP  32
#define HDIM  1024
#define IDIM  768

// Tiling
#define BM 32                 // tokens per workgroup tile
#define HS (IDIM + 8)         // padded LDS row stride (bf16 elems) to dodge bank conflicts

typedef __attribute__((ext_vector_type(16))) __bf16 v16bf;
typedef __attribute__((ext_vector_type(8)))  float  v8f;

// Fragment viewed either as the WMMA operand vector or as two 16B quads
// (for direct b128 loads of pre-converted bf16 data).
union FragBF {
  v16bf v;
  uint4 q[2];
};

// ---- fragment loaders -------------------------------------------------------
// Load 8 contiguous fp32 from global, convert to bf16 into frag[o..o+7]
// (compiler pairs these into v_cvt_pk_bf16_f32).
__device__ __forceinline__ void ldg_half8(const float* __restrict__ p, v16bf& f, int o) {
  const float4* q = (const float4*)p;
  float4 a = q[0];
  float4 b = q[1];
  f[o + 0] = (__bf16)a.x; f[o + 1] = (__bf16)a.y; f[o + 2] = (__bf16)a.z; f[o + 3] = (__bf16)a.w;
  f[o + 4] = (__bf16)b.x; f[o + 5] = (__bf16)b.y; f[o + 6] = (__bf16)b.z; f[o + 7] = (__bf16)b.w;
}

// Load 16 contiguous fp32 from global -> bf16 fragment (B-matrix per-lane chunk).
__device__ __forceinline__ v16bf ldg_frag16(const float* __restrict__ p) {
  v16bf f;
  ldg_half8(p, f, 0);
  ldg_half8(p + 8, f, 8);
  return f;
}

// ---- kernel 0: zero output / combine / counters -----------------------------
__global__ void zero_kernel(float* __restrict__ out, float* __restrict__ combine,
                            int* __restrict__ cnt) {
  int i = blockIdx.x * blockDim.x + threadIdx.x;
  if (i < T_TOK * HDIM) out[i] = 0.0f;
  if (i < T_TOK * NEXP) combine[i] = 0.0f;
  if (i < NEXP) cnt[i] = 0;
}

// ---- kernel 0b: pre-convert hidden_states fp32 -> bf16 (one cheap pass) -----
__global__ void cvt_x_kernel(const float* __restrict__ X, uint4* __restrict__ xb) {
  int i = blockIdx.x * blockDim.x + threadIdx.x;   // one uint4 = 8 bf16 per thread
  if (i >= T_TOK * HDIM / 8) return;
  FragBF f;
  ldg_half8(X + (size_t)i * 8, f.v, 0);
  xb[i] = f.q[0];
}

// ---- kernel 1: scatter top-k weights into dense combine[t][e] ---------------
__global__ void scatter_kernel(const int* __restrict__ idx, const float* __restrict__ w,
                               float* __restrict__ combine) {
  int i = blockIdx.x * blockDim.x + threadIdx.x;
  if (i >= T_TOK * KSEL) return;
  int t = i / KSEL;
  atomicAdd(&combine[t * NEXP + idx[i]], w[i]);
}

// ---- kernel 2: deterministic per-expert token-list compaction ---------------
__global__ __launch_bounds__(256) void build_lists(const float* __restrict__ combine,
                                                   int* __restrict__ cnt,
                                                   int* __restrict__ list,
                                                   float* __restrict__ wlist) {
  __shared__ int pref[256];
  __shared__ int base;
  const int e = blockIdx.x;
  const int tid = threadIdx.x;
  if (tid == 0) base = 0;
  __syncthreads();
  for (int t0 = 0; t0 < T_TOK; t0 += 256) {
    int t = t0 + tid;
    float w = combine[t * NEXP + e];
    int p = (w != 0.0f) ? 1 : 0;
    pref[tid] = p;
    __syncthreads();
    // inclusive Hillis-Steele scan over 256 entries
    for (int off = 1; off < 256; off <<= 1) {
      int v = (tid >= off) ? pref[tid - off] : 0;
      __syncthreads();
      pref[tid] += v;
      __syncthreads();
    }
    if (p) {
      int pos = base + pref[tid] - 1;
      list[e * T_TOK + pos] = t;
      wlist[e * T_TOK + pos] = w;
    }
    __syncthreads();
    if (tid == 0) base += pref[255];
    __syncthreads();
  }
  if (tid == 0) cnt[e] = base;
}

// ---- kernel 3: fused SwiGLU MoE expert GEMMs (bf16 WMMA, fp32 accum) --------
// grid = (ceil(T/BM), NEXP); block = 256 threads = 8 waves laid out 2(M) x 4(N).
__global__ __launch_bounds__(256) void moe_fused_kernel(
    const __bf16* __restrict__ Xb,  // [T, H] pre-converted bf16
    const float* __restrict__ Wg,   // [E, I, H]
    const float* __restrict__ Wu,   // [E, I, H]
    const float* __restrict__ Wd,   // [E, H, I]
    const int*   __restrict__ list, // [E, T]
    const float* __restrict__ wlist,// [E, T]
    const int*   __restrict__ cnt,  // [E]
    float*       __restrict__ out)  // [T, H]
{
  __shared__ __bf16 h[BM * HS];   // intermediate activations, bf16
  __shared__ int    tl[BM];
  __shared__ float  wl[BM];

  const int e   = blockIdx.y;
  const int blk = blockIdx.x;
  const int ce  = cnt[e];
  if (blk * BM >= ce) return;      // uniform early exit (no barriers crossed)

  const int tid  = threadIdx.x;
  const int lane = tid & 31;
  const int wid  = tid >> 5;
  const int wm   = wid & 1;        // M sub-block (16 rows each)
  const int wn   = wid >> 1;       // N sub-block (16 cols each, 4 waves wide)
  const int half = lane >> 4;      // K-half selector per ISA 16-bit layouts
  const int l15  = lane & 15;

  if (tid < BM) {
    int idx = blk * BM + tid;
    int ok = idx < ce;
    tl[tid] = list[e * T_TOK + (ok ? idx : 0)];   // clamp to a valid token
    wl[tid] = ok ? wlist[e * T_TOK + idx] : 0.0f; // zero weight kills tail rows
  }
  __syncthreads();

  const float* Ge = Wg + (size_t)e * IDIM * HDIM;
  const float* Ue = Wu + (size_t)e * IDIM * HDIM;
  const float* De = Wd + (size_t)e * HDIM * IDIM;

  const int trow = tl[wm * 16 + l15];   // token row this lane feeds into A-frag
  const __bf16* xrow = Xb + (size_t)trow * HDIM;

  // ---------------- stage 1: h = silu(X Ge^T) * (X Ue^T) ----------------
#pragma unroll 1
  for (int n0 = 0; n0 < IDIM; n0 += 64) {
    const int ncol = n0 + wn * 16 + l15;     // row of Ge/Ue (== output column)
    v8f accg = {};
    v8f accu = {};
#pragma unroll 2
    for (int kb = 0; kb < HDIM; kb += 32) {
      // A fragment: pre-converted bf16 X row, two 16B quads (no cvt in loop)
      FragBF a;
      const __bf16* xa = xrow + kb + half * 16;
      a.q[0] = *(const uint4*)(xa);
      a.q[1] = *(const uint4*)(xa + 8);
      // B fragments: contiguous K-chunk of the fp32 weight row -> bf16
      const float* gp = Ge + (size_t)ncol * HDIM + kb + half * 16;
      const float* up = Ue + (size_t)ncol * HDIM + kb + half * 16;
      __builtin_prefetch(gp + 64, 0, 1);     // stream-ahead on HBM weights
      __builtin_prefetch(up + 64, 0, 1);
      v16bf bg = ldg_frag16(gp);
      v16bf bu = ldg_frag16(up);
      accg = __builtin_amdgcn_wmma_f32_16x16x32_bf16(false, a.v, false, bg,
                                                     (short)0, accg, false, false);
      accu = __builtin_amdgcn_wmma_f32_16x16x32_bf16(false, a.v, false, bu,
                                                     (short)0, accu, false, false);
    }
#pragma unroll
    for (int r = 0; r < 8; r++) {
      float g = accg[r];
      float u = accu[r];
      // silu(g)*u with hardware v_rcp_f32 instead of IEEE divide chain
      float s  = __builtin_amdgcn_rcpf(1.0f + __expf(-g));
      float hv = g * s * u;
      int row = wm * 16 + r + half * 8;             // C-layout: M = r (+8 for hi lanes)
      int col = n0 + wn * 16 + l15;
      h[row * HS + col] = (__bf16)hv;
    }
  }
  __syncthreads();

  // Hoist per-row token ids / combine weights out of the n0 loop (regs, not LDS).
  int   tok[8];
  float cw[8];
#pragma unroll
  for (int r = 0; r < 8; r++) {
    int slot = wm * 16 + r + half * 8;
    tok[r] = tl[slot];
    cw[r]  = wl[slot];
  }

  // ---------------- stage 2: out += (h Wd^T) * combine_w ----------------
  const __bf16* hrow = &h[(wm * 16 + l15) * HS];   // this lane's A-matrix row
#pragma unroll 1
  for (int n0 = 0; n0 < HDIM; n0 += 64) {
    const int nrow = n0 + wn * 16 + l15;     // row of De (== output column)
    v8f acc = {};
#pragma unroll 2
    for (int kb = 0; kb < IDIM; kb += 32) {
      // A fragment straight from LDS as two 16B quads (ds_load_b128)
      FragBF a;
      const __bf16* ha = hrow + kb + half * 16;
      a.q[0] = *(const uint4*)(ha);
      a.q[1] = *(const uint4*)(ha + 8);
      const float* dp = De + (size_t)nrow * IDIM + kb + half * 16;
      __builtin_prefetch(dp + 64, 0, 1);
      v16bf b = ldg_frag16(dp);
      acc = __builtin_amdgcn_wmma_f32_16x16x32_bf16(false, a.v, false, b,
                                                    (short)0, acc, false, false);
    }
#pragma unroll
    for (int r = 0; r < 8; r++) {
      atomicAdd(&out[(size_t)tok[r] * HDIM + n0 + wn * 16 + l15], acc[r] * cw[r]);
    }
  }
}

// ---- launcher ---------------------------------------------------------------
extern "C" void kernel_launch(void* const* d_in, const int* in_sizes, int n_in,
                              void* d_out, int out_size, void* d_ws, size_t ws_size,
                              hipStream_t stream) {
  const float* X    = (const float*)d_in[0];  // hidden_states [T,H]
  const int*   tki  = (const int*)  d_in[1];  // top_k_index [T,K]
  const float* tkw  = (const float*)d_in[2];  // top_k_weights [T,K]
  const float* Wg   = (const float*)d_in[3];  // gate_proj [E,I,H]
  const float* Wu   = (const float*)d_in[4];  // up_proj [E,I,H]
  const float* Wd   = (const float*)d_in[5];  // down_proj [E,H,I]
  float* out = (float*)d_out;

  // workspace layout (~4.8 MB total)
  float*  combine = (float*)d_ws;                      // T*E fp32
  int*    cnt     = (int*)(combine + T_TOK * NEXP);    // E
  int*    list    = cnt + NEXP;                        // E*T
  float*  wlist   = (float*)(list + NEXP * T_TOK);     // E*T
  __bf16* xbf     = (__bf16*)(wlist + NEXP * T_TOK);   // T*H bf16 (16B-aligned offset)

  zero_kernel<<<(T_TOK * HDIM + 255) / 256, 256, 0, stream>>>(out, combine, cnt);
  cvt_x_kernel<<<(T_TOK * HDIM / 8 + 255) / 256, 256, 0, stream>>>(X, (uint4*)xbf);
  scatter_kernel<<<(T_TOK * KSEL + 255) / 256, 256, 0, stream>>>(tki, tkw, combine);
  build_lists<<<NEXP, 256, 0, stream>>>(combine, cnt, list, wlist);

  dim3 grid(T_TOK / BM, NEXP);   // 64 token-tiles x 32 experts; x-fastest keeps
                                 // same-expert tiles adjacent for L2 weight reuse
  moe_fused_kernel<<<grid, 256, 0, stream>>>(xbf, Wg, Wu, Wd, list, wlist, cnt, out);
}
